// AttentionEntropyEstimator_71545565217047
// MI455X (gfx1250) — matrix operand order repaired and checked by reading, not compiled, not run on hardware
//
#include <hip/hip_runtime.h>
#include <hip/hip_bf16.h>
#include <math.h>
#include <stdint.h>

// ---------------------------------------------------------------------------
// AttentionEntropyEstimator for MI455X (gfx1250, wave32, WMMA + async LDS)
// B=4, L=2048, D=1024, H=4, hd=256.  All big math in bf16 WMMA w/ f32 acc.
// ---------------------------------------------------------------------------

typedef __attribute__((ext_vector_type(16))) __bf16 v16bf;
typedef __attribute__((ext_vector_type(8)))  __bf16 v8bf;
typedef __attribute__((ext_vector_type(4)))  __bf16 v4bf;
typedef __attribute__((ext_vector_type(8)))  float  v8f;
typedef __attribute__((ext_vector_type(4)))  float  v4f;

#define D_MODEL 1024
#define QK_LD   2048          // combined [Q | K] row length (bf16 elements)
#define B_SZ    4
#define L_SEQ   2048
#define NHEAD   4
#define HDIM    256
#define SCALE   (1.0f/16.0f)  // 1/sqrt(256)

// K-tile staging in LDS (score kernels): 64 s-rows x 256 d-cols bf16.
// Row stride 528 B = 132 DWORDs -> lane bank stride 4 -> conflict-free b128.
#define KT_ROWS     64
#define KT_STRIDE_B 528
#define KT_BUF_B    (KT_ROWS * KT_STRIDE_B)   // 33792 B per buffer
#define KT_NT       (L_SEQ / KT_ROWS)         // 32 tiles

// A fragment (16x32 bf16, M x K), src row-major [row][k], ld in elements.
__device__ __forceinline__ v16bf load_a_frag(const __bf16* __restrict__ base,
                                             int ld, int row, int k0, int lane) {
  const int half = lane >> 4;
  const int m    = lane & 15;
  const __bf16* p = base + (size_t)(row + m) * ld + k0 + half * 16;
  union { v16bf v; v8bf h[2]; } u;
  u.h[0] = *(const v8bf*)(p);
  u.h[1] = *(const v8bf*)(p + 8);
  return u.v;
}

// B fragment (32x16 bf16, K x N), logical B[k][n] = src[n][k], src row-major
// N x K (ld = K stride): lane n holds column n, one contiguous 32B read.
__device__ __forceinline__ v16bf load_b_frag(const __bf16* __restrict__ base,
                                             int ld, int col, int k0, int lane) {
  const int half = lane >> 4;
  const int n    = lane & 15;
  const __bf16* p = base + (size_t)(col + n) * ld + k0 + half * 16;
  return *(const v16bf*)(p);
}

// B fragment from the padded LDS K-tile. row = sub*16 + lane%16,
// byte col = kk*64 + half*32; two 16B-aligned halves.
__device__ __forceinline__ v16bf lds_b_frag(const unsigned char* lbase,
                                            int sub, int kk, int lane) {
  const int half = lane >> 4;
  const int n    = lane & 15;
  const unsigned char* p = lbase + (sub * 16 + n) * KT_STRIDE_B
                                 + kk * 64 + half * 32;
  union { v16bf v; v8bf h[2]; } u;
  u.h[0] = *(const v8bf*)(p);
  u.h[1] = *(const v8bf*)(p + 16);
  return u.v;
}

__device__ __forceinline__ v8f wmma_bf16(v16bf a, v16bf b, v8f c) {
  return __builtin_amdgcn_wmma_f32_16x16x32_bf16(false, a, false, b,
                                                 (short)0, c, false, false);
}

// Stage one 64-row x 512B K tile into LDS via async copy (ASYNCcnt path).
// 2048 b128 chunks / 256 threads = 8 per thread.
__device__ __forceinline__ void stage_ktile(const __bf16* __restrict__ gsrc,
                                            unsigned lds0, int tid) {
#pragma unroll
  for (int i = 0; i < 8; ++i) {
    const int c   = tid + (i << 8);        // 0..2047
    const int row = c >> 5;
    const int cir = c & 31;
    unsigned long long g = (unsigned long long)(uintptr_t)gsrc
                         + (unsigned long long)row * (QK_LD * 2)
                         + (unsigned)(cir << 4);
    unsigned l = lds0 + (unsigned)row * KT_STRIDE_B + (unsigned)(cir << 4);
    asm volatile("global_load_async_to_lds_b128 %0, %1, off"
                 :: "v"(l), "v"(g) : "memory");
  }
}

__device__ __forceinline__ void wait_async0() {
  asm volatile("s_wait_asynccnt 0" ::: "memory");
}

// ---------------------------------------------------------------------------
__global__ void k_f32_to_bf16(const float* __restrict__ in,
                              __bf16* __restrict__ out, int n) {
  int i = (blockIdx.x * blockDim.x + threadIdx.x) * 4;
  if (i + 3 < n) {
    v4f x = *(const v4f*)(in + i);
    v4bf y;
    y[0] = (__bf16)x[0]; y[1] = (__bf16)x[1];
    y[2] = (__bf16)x[2]; y[3] = (__bf16)x[3];
    *(v4bf*)(out + i) = y;
  }
}

__global__ void k_zero(float* __restrict__ p, int n) {
  int i = blockIdx.x * blockDim.x + threadIdx.x;
  if (i < n) p[i] = 0.0f;
}

// ---------------------------------------------------------------------------
// Fused Q|K projection: qk[m][n] = sum_d hs[m][d]*w[n][d] + bias[n]
// M=8192, N=2048 (cols 0..1023 = Q, 1024..2047 = K), K=1024.
__global__ void __launch_bounds__(256)
k_proj_gemm(const __bf16* __restrict__ hs, const __bf16* __restrict__ w,
            const float* __restrict__ bias, __bf16* __restrict__ qk) {
  const int lane = threadIdx.x & 31;
  const int wid  = threadIdx.x >> 5;
  const int wm = wid & 1, wn = wid >> 1;
  const int row0 = blockIdx.x * 64  + wm * 32;
  const int col0 = blockIdx.y * 128 + wn * 32;

  const v8f vzero = {0.f,0.f,0.f,0.f,0.f,0.f,0.f,0.f};
  v8f acc[2][2];
#pragma unroll
  for (int i = 0; i < 2; ++i)
#pragma unroll
    for (int j = 0; j < 2; ++j) acc[i][j] = vzero;

  for (int k0 = 0; k0 < D_MODEL; k0 += 32) {
    v16bf a0 = load_a_frag(hs, D_MODEL, row0,      k0, lane);
    v16bf a1 = load_a_frag(hs, D_MODEL, row0 + 16, k0, lane);
    v16bf b0 = load_b_frag(w,  D_MODEL, col0,      k0, lane);
    v16bf b1 = load_b_frag(w,  D_MODEL, col0 + 16, k0, lane);
    acc[0][0] = wmma_bf16(a0, b0, acc[0][0]);
    acc[0][1] = wmma_bf16(a0, b1, acc[0][1]);
    acc[1][0] = wmma_bf16(a1, b0, acc[1][0]);
    acc[1][1] = wmma_bf16(a1, b1, acc[1][1]);
  }

  const int half = lane >> 4, nn = lane & 15;
#pragma unroll
  for (int i = 0; i < 2; ++i) {
#pragma unroll
    for (int j = 0; j < 2; ++j) {
      const int col = col0 + j * 16 + nn;
      const float bv = bias[col];
#pragma unroll
      for (int v = 0; v < 8; ++v) {
        const int rowm = row0 + i * 16 + half * 8 + v;
        qk[(size_t)rowm * QK_LD + col] = (__bf16)(acc[i][j][v] + bv);
      }
    }
  }
}

// ---------------------------------------------------------------------------
// Pass 1: per-row (b,h,l) online softmax stats (max, sum-exp) over s.
// 256 blocks = (b,h,l-tile-of-128). K tiles double-buffered in LDS via
// global_load_async_to_lds_b128; lane-local online (m,s), merged once at end.
__global__ void __launch_bounds__(256)
k_scores_stats(const __bf16* __restrict__ qk,
               float* __restrict__ smax, float* __restrict__ ssum) {
  __shared__ __align__(16) unsigned char ktile[2][KT_BUF_B];

  const int tid  = threadIdx.x;
  const int lane = tid & 31;
  const int wid  = tid >> 5;
  const int blk  = blockIdx.x;
  const int b  = blk >> 6;
  const int h  = (blk >> 4) & 3;
  const int lt = blk & 15;
  const int l0 = lt * 128 + wid * 16;

  const __bf16* qbase = qk + (size_t)b * L_SEQ * QK_LD + (size_t)h * HDIM;
  const __bf16* kbase = qbase + 1024;

  v16bf afr[8];
#pragma unroll
  for (int kk = 0; kk < 8; ++kk) afr[kk] = load_a_frag(qbase, QK_LD, l0, kk * 32, lane);

  const unsigned lds0 = (unsigned)(uintptr_t)&ktile[0][0];
  const unsigned lds1 = (unsigned)(uintptr_t)&ktile[1][0];

  const v8f vzero = {0.f,0.f,0.f,0.f,0.f,0.f,0.f,0.f};
  float m[8], s[8];
#pragma unroll
  for (int v = 0; v < 8; ++v) { m[v] = -3.0e38f; s[v] = 0.f; }

  stage_ktile(kbase, lds0, tid);                       // tile 0 -> buf 0
  for (int t = 0; t < KT_NT; ++t) {
    wait_async0();                                     // own chunks landed
    __syncthreads();                                   // everyone's landed
    if (t + 1 < KT_NT)
      stage_ktile(kbase + (size_t)(t + 1) * KT_ROWS * QK_LD,
                  ((t + 1) & 1) ? lds1 : lds0, tid);

    const unsigned char* lbase = ktile[t & 1];
#pragma unroll
    for (int sub = 0; sub < 4; ++sub) {
      v8f acc = vzero;
#pragma unroll
      for (int kk = 0; kk < 8; ++kk)
        acc = wmma_bf16(afr[kk], lds_b_frag(lbase, sub, kk, lane), acc);
#pragma unroll
      for (int v = 0; v < 8; ++v) {                    // lane-local online
        const float val = acc[v] * SCALE;
        const float nm  = fmaxf(m[v], val);
        s[v] = s[v] * __expf(m[v] - nm) + __expf(val - nm);
        m[v] = nm;
      }
    }
  }

  // merge (m,s) across the 16 columns of each half, once
#pragma unroll
  for (int v = 0; v < 8; ++v) {
#pragma unroll
    for (int mk = 1; mk < 16; mk <<= 1) {
      const float om = __shfl_xor(m[v], mk, 32);
      const float os = __shfl_xor(s[v], mk, 32);
      const float nm = fmaxf(m[v], om);
      s[v] = s[v] * __expf(m[v] - nm) + os * __expf(om - nm);
      m[v] = nm;
    }
  }

  if ((lane & 15) == 0) {
    const int half = lane >> 4;                        // rows half*8 .. +7
    const size_t base = ((size_t)(b * NHEAD + h)) * L_SEQ + l0 + half * 8;
#pragma unroll
    for (int v = 0; v < 8; ++v) { smax[base + v] = m[v]; ssum[base + v] = s[v]; }
  }
}

// ---------------------------------------------------------------------------
// Pass 2: recompute scores from LDS-staged K tiles, p = exp(z-m)/sum,
// column-sum into LDS partial, one global-atomic flush per block.
__global__ void __launch_bounds__(256)
k_scores_accum(const __bf16* __restrict__ qk,
               const float* __restrict__ smax, const float* __restrict__ ssum,
               float* __restrict__ aw) {
  __shared__ float awp[L_SEQ];
  __shared__ __align__(16) unsigned char ktile[2][KT_BUF_B];

  const int tid  = threadIdx.x;
  for (int i = tid; i < L_SEQ; i += 256) awp[i] = 0.f;

  const int lane = tid & 31;
  const int wid  = tid >> 5;
  const int blk  = blockIdx.x;
  const int b  = blk >> 6;
  const int h  = (blk >> 4) & 3;
  const int lt = blk & 15;
  const int l0 = lt * 128 + wid * 16;

  const __bf16* qbase = qk + (size_t)b * L_SEQ * QK_LD + (size_t)h * HDIM;
  const __bf16* kbase = qbase + 1024;

  v16bf afr[8];
#pragma unroll
  for (int kk = 0; kk < 8; ++kk) afr[kk] = load_a_frag(qbase, QK_LD, l0, kk * 32, lane);

  const int half = lane >> 4;
  const size_t sbase = ((size_t)(b * NHEAD + h)) * L_SEQ + l0 + half * 8;
  float m[8], is[8];
#pragma unroll
  for (int v = 0; v < 8; ++v) {
    m[v]  = smax[sbase + v];
    is[v] = 1.0f / ssum[sbase + v];
  }

  const unsigned lds0 = (unsigned)(uintptr_t)&ktile[0][0];
  const unsigned lds1 = (unsigned)(uintptr_t)&ktile[1][0];
  const v8f vzero = {0.f,0.f,0.f,0.f,0.f,0.f,0.f,0.f};

  stage_ktile(kbase, lds0, tid);
  for (int t = 0; t < KT_NT; ++t) {
    wait_async0();
    __syncthreads();
    if (t + 1 < KT_NT)
      stage_ktile(kbase + (size_t)(t + 1) * KT_ROWS * QK_LD,
                  ((t + 1) & 1) ? lds1 : lds0, tid);

    const unsigned char* lbase = ktile[t & 1];
#pragma unroll
    for (int sub = 0; sub < 4; ++sub) {
      v8f acc = vzero;
#pragma unroll
      for (int kk = 0; kk < 8; ++kk)
        acc = wmma_bf16(afr[kk], lds_b_frag(lbase, sub, kk, lane), acc);

      float colsum = 0.f;
#pragma unroll
      for (int v = 0; v < 8; ++v)
        colsum += __expf(acc[v] * SCALE - m[v]) * is[v];
      colsum += __shfl_xor(colsum, 16, 32);            // combine both halves
      if (half == 0)
        atomicAdd(&awp[t * KT_ROWS + sub * 16 + (lane & 15)], colsum);
    }
  }
  __syncthreads();

  float* awg = aw + (size_t)b * L_SEQ;
  for (int i = tid; i < L_SEQ; i += 256) atomicAdd(&awg[i], awp[i]);
}

// ---------------------------------------------------------------------------
// Finalize: aw_sum -> probs (/H/L + eps), entropy, mean over b, sigmoid.
__global__ void __launch_bounds__(256)
k_finalize(const float* __restrict__ aw, float* __restrict__ out) {
  __shared__ float red[256];
  float local = 0.f;
  const float inv = 1.0f / (float)(NHEAD * L_SEQ);
  for (int i = threadIdx.x; i < B_SZ * L_SEQ; i += 256) {
    const float a = aw[i] * inv + 1e-8f;
    local += -a * __logf(a);
  }
  red[threadIdx.x] = local;
  __syncthreads();
  for (int sft = 128; sft > 0; sft >>= 1) {
    if (threadIdx.x < sft) red[threadIdx.x] += red[threadIdx.x + sft];
    __syncthreads();
  }
  if (threadIdx.x == 0) {
    const float me = red[0] * (1.0f / (float)B_SZ);
    out[0] = 1.0f / (1.0f + __expf(-me));
  }
}

// ---------------------------------------------------------------------------
extern "C" void kernel_launch(void* const* d_in, const int* in_sizes, int n_in,
                              void* d_out, int out_size, void* d_ws, size_t ws_size,
                              hipStream_t stream) {
  (void)in_sizes; (void)n_in; (void)out_size; (void)ws_size;
  const float* hs  = (const float*)d_in[0];   // (4,2048,1024)
  const float* win = (const float*)d_in[1];   // (3072,1024), rows 0..2047 used
  const float* bin = (const float*)d_in[2];   // (3072,), first 2048 used
  float* out = (float*)d_out;

  char* ws = (char*)d_ws;
  __bf16* hsb = (__bf16*)ws;  ws += (size_t)8192 * 1024 * 2;   // 16 MB
  __bf16* wb  = (__bf16*)ws;  ws += (size_t)2048 * 1024 * 2;   //  4 MB
  __bf16* qkb = (__bf16*)ws;  ws += (size_t)8192 * 2048 * 2;   // 32 MB
  float* smax = (float*)ws;   ws += (size_t)32768 * 4;
  float* ssum = (float*)ws;   ws += (size_t)32768 * 4;
  float* aw   = (float*)ws;   ws += (size_t)8192 * 4;

  k_f32_to_bf16<<<8192, 256, 0, stream>>>(hs,  hsb, 8192 * 1024);
  k_f32_to_bf16<<<2048, 256, 0, stream>>>(win, wb,  2048 * 1024);
  k_zero<<<32, 256, 0, stream>>>(aw, 8192);

  dim3 gProj(128, 16);
  k_proj_gemm<<<gProj, 256, 0, stream>>>(hsb, wb, bin, qkb);

  k_scores_stats<<<256, 256, 0, stream>>>(qkb, smax, ssum);
  k_scores_accum<<<256, 256, 0, stream>>>(qkb, smax, ssum, aw);
  k_finalize<<<1, 256, 0, stream>>>(aw, out);
}
